// AttentionHistogram_75453985457226
// MI455X (gfx1250) — compile-verified, matrix operand order checked by reading, not run on hardware
//
#include <hip/hip_runtime.h>
#include <stdint.h>

// ---------------------------------------------------------------- constants
constexpr int kB     = 4;
constexpr int kC     = 256;   // DIM
constexpr int kHalf  = 128;
constexpr int kHeads = 8;
constexpr int kH     = 96;
constexpr int kW     = 96;
constexpr int kN     = kH * kW;        // 9216 pixels
constexpr int kQKV   = kC * 5;         // 1280
constexpr int kHW    = kN / kHeads;    // 1152 (attention d)
constexpr int kM     = 256;            // cph * factor (attention rows)

// ---------------------------------------------------------------- types
typedef __attribute__((ext_vector_type(16))) __bf16       v16bf;
typedef __attribute__((ext_vector_type(8)))  float        v8f;
typedef __attribute__((ext_vector_type(4)))  unsigned int u32x4;
typedef __attribute__((ext_vector_type(8)))  int          i32x8;
typedef __attribute__((ext_vector_type(4)))  int          i32x4;

// ---------------------------------------------------------------- helpers
__device__ inline v8f zero8() {
  v8f z;
#pragma unroll
  for (int i = 0; i < 8; ++i) z[i] = 0.0f;
  return z;
}

__device__ inline v8f wmma_bf16(v16bf a, v16bf b, v8f c) {
  // D = A(16x32) * B(32x16) + C, fp32 accumulate
  return __builtin_amdgcn_wmma_f32_16x16x32_bf16(false, a, false, b,
                                                 (short)0, c, false, false);
}

// LDS byte offset of a __shared__ pointer (generic addr low 32 bits = LDS offset)
__device__ inline unsigned lds_byte_off(const void* p) {
  return (unsigned)(uintptr_t)p;
}

// Tensor Data Mover: 2D bf16 tile (tile_rows x tile_cols), row-major source with
// row stride `stride_elems`, packed contiguously into LDS at lds_off.
// clang-23 / therock-10.0 6-arg builtin; D# packed per CDNA5 ISA.
__device__ inline void tdm_load_2d_bf16(unsigned lds_off, const __bf16* g,
                                        unsigned tile_cols, unsigned tile_rows,
                                        unsigned stride_elems) {
  unsigned long long ga = (unsigned long long)(uintptr_t)g;
  u32x4 g0;
  g0[0] = 1u;                                           // count=1 (valid), user mode
  g0[1] = lds_off;                                      // lds_addr
  g0[2] = (unsigned)ga;                                 // global_addr[31:0]
  g0[3] = (unsigned)((ga >> 32) & 0x01FFFFFFu) | (2u << 30);  // addr hi | type=2
  i32x8 g1;
  g1[0] = 1 << 16;                                      // data_size=1 -> 2 bytes
  g1[1] = (int)((tile_cols & 0xFFFFu) << 16);           // tensor_dim0 lo16
  g1[2] = (int)((tile_cols >> 16) & 0xFFFFu) |          // tensor_dim0 hi16
          (int)((tile_rows & 0xFFFFu) << 16);           // tensor_dim1 lo16
  g1[3] = (int)((tile_rows >> 16) & 0xFFFFu) |          // tensor_dim1 hi16
          (int)((tile_cols & 0xFFFFu) << 16);           // tile_dim0
  g1[4] = (int)(tile_rows & 0xFFFFu);                   // tile_dim1 (tile_dim2=0)
  g1[5] = (int)stride_elems;                            // tensor_dim0_stride[31:0]
  g1[6] = 0;
  g1[7] = 0;
  i32x4 gz;
  gz[0] = 0; gz[1] = 0; gz[2] = 0; gz[3] = 0;
  i32x8 gz8;
#pragma unroll
  for (int i = 0; i < 8; ++i) gz8[i] = 0;
  __builtin_amdgcn_tensor_load_to_lds(g0, g1, gz, gz, gz8, 0);
}

// A fragment: 16x32 bf16 tile, row-major in LDS with leading dim ld.
// ISA §7.12.2 16-bit A layout: lane L row=L%16; half (L/16) selects K offset 8;
// VGPR v holds K pairs {0..7,16..23} (+half*8).
__device__ inline v16bf load_a_frag(const __bf16* p, int ld) {
  const int lane = threadIdx.x & 31;
  const __bf16* r = p + (lane & 15) * ld;
  const int koff = (lane >> 4) * 8;
  v16bf f;
#pragma unroll
  for (int i = 0; i < 8; ++i) {
    int kk = ((i < 4) ? (2 * i) : (16 + 2 * (i - 4))) + koff;
    f[2 * i]     = r[kk];
    f[2 * i + 1] = r[kk + 1];
  }
  return f;
}

// B fragment: 32x16 (KxN) from LDS tile stored K-major (row stride ld over N).
__device__ inline v16bf load_b_frag(const __bf16* p, int ld) {
  const int lane = threadIdx.x & 31;
  const int col  = lane & 15;
  const int koff = (lane >> 4) * 8;
  v16bf f;
#pragma unroll
  for (int i = 0; i < 8; ++i) {
    int kk = ((i < 4) ? (2 * i) : (16 + 2 * (i - 4))) + koff;
    f[2 * i]     = p[(size_t)kk * ld + col];
    f[2 * i + 1] = p[(size_t)(kk + 1) * ld + col];
  }
  return f;
}

// C store: VGPR r -> row r + 8*(lane/16), col = lane%16.
__device__ inline void store_c(float* p, int ldc, v8f c) {
  const int lane = threadIdx.x & 31;
  const int col  = lane & 15;
  const int rb   = (lane >> 4) * 8;
#pragma unroll
  for (int r = 0; r < 8; ++r) p[(size_t)(r + rb) * ldc + col] = c[r];
}

// 128x128 block GEMM: C = A(128xK) * B(Kx128), bf16 in, fp32 out.
// 256 threads = 8 waves tiled 2(M) x 4(N); TDM feeds LDS, wave0 issues.
__device__ inline void gemm128x128_bf16(const __bf16* A, int lda,
                                        const __bf16* Bm, int ldb,
                                        float* C, int ldc, int K) {
  __shared__ __bf16 lA[128 * 32];
  __shared__ __bf16 lB[32 * 128];
  const int wave = threadIdx.x >> 5;
  const int wm = wave >> 2;     // 0..1 -> 64 rows
  const int wn = wave & 3;      // 0..3 -> 32 cols
  v8f acc[4][2];
#pragma unroll
  for (int i = 0; i < 4; ++i)
#pragma unroll
    for (int j = 0; j < 2; ++j) acc[i][j] = zero8();

  for (int k0 = 0; k0 < K; k0 += 32) {
    __syncthreads();
    if (threadIdx.x < 32) {   // wave0 drives the TDM; EXEC ignored by tensor ops
      tdm_load_2d_bf16(lds_byte_off(lA), A + k0, 32u, 128u, (unsigned)lda);
      tdm_load_2d_bf16(lds_byte_off(lB), Bm + (size_t)k0 * ldb, 128u, 32u,
                       (unsigned)ldb);
      __builtin_amdgcn_s_wait_tensorcnt(0);
    }
    __syncthreads();
    const __bf16* pa = lA + (size_t)(wm * 64) * 32;
    const __bf16* pb = lB + wn * 32;
    v16bf bf0 = load_b_frag(pb, 128);
    v16bf bf1 = load_b_frag(pb + 16, 128);
#pragma unroll
    for (int i = 0; i < 4; ++i) {
      v16bf af = load_a_frag(pa + (size_t)(i * 16) * 32, 32);
      acc[i][0] = wmma_bf16(af, bf0, acc[i][0]);
      acc[i][1] = wmma_bf16(af, bf1, acc[i][1]);
    }
  }
#pragma unroll
  for (int i = 0; i < 4; ++i)
#pragma unroll
    for (int j = 0; j < 2; ++j)
      store_c(C + (size_t)(wm * 64 + i * 16) * ldc + (wn * 32 + j * 16), ldc,
              acc[i][j]);
}

// ---------------------------------------------------------------- kernel 1:
// stable sort of x[:, :128] along H then W; emit bf16 x for the qkv GEMM.
__device__ inline void isort96(float* v, int* id) {
  for (int i = 1; i < 96; ++i) {
    float key = v[i];
    int kid = id[i];
    int j = i - 1;
    while (j >= 0 && v[j] > key) { v[j + 1] = v[j]; id[j + 1] = id[j]; --j; }
    v[j + 1] = key; id[j + 1] = kid;
  }
}

__global__ void __launch_bounds__(96) k_sort_hw(const float* __restrict__ x,
                                                __bf16* __restrict__ x_bf,
                                                int* __restrict__ idx_h,
                                                int* __restrict__ idx_w) {
  __shared__ float tile[96 * 96];
  const int c = blockIdx.x, b = blockIdx.y, t = threadIdx.x;
  const size_t base = ((size_t)b * kC + c) * kN;
  if (c >= kHalf) {
    for (int w = 0; w < 96; ++w) {
      size_t o = (size_t)t * 96 + w;
      x_bf[base + o] = (__bf16)x[base + o];
    }
    return;
  }
  const size_t ib = ((size_t)b * kHalf + c) * kN;
  float pv[96];
  int pid[96];
  // column sort (thread t owns column w=t)
  for (int h = 0; h < 96; ++h) { pv[h] = x[base + (size_t)h * 96 + t]; pid[h] = h; }
  isort96(pv, pid);
  for (int h = 0; h < 96; ++h) {
    tile[h * 96 + t] = pv[h];
    idx_h[ib + (size_t)h * 96 + t] = pid[h];
  }
  __syncthreads();
  // row sort (thread t owns row h=t)
  for (int w = 0; w < 96; ++w) { pv[w] = tile[t * 96 + w]; pid[w] = w; }
  isort96(pv, pid);
  for (int w = 0; w < 96; ++w) {
    x_bf[base + (size_t)t * 96 + w] = (__bf16)pv[w];
    idx_w[ib + (size_t)t * 96 + w] = pid[w];
  }
}

// ---------------------------------------------------------------- kernel 2:
// weight fp32 -> bf16
__global__ void k_cvt_w(const float* __restrict__ wq, const float* __restrict__ wp,
                        __bf16* __restrict__ wqb, __bf16* __restrict__ wpb) {
  const int i = blockIdx.x * 256 + threadIdx.x;
  if (i < kQKV * kC) wqb[i] = (__bf16)wq[i];
  if (i < kC * kC) wpb[i] = (__bf16)wp[i];
}

// ---------------------------------------------------------------- kernel 3:
// qkv 1x1 conv as GEMM: (1280x256) x (256x9216) per batch
__global__ void __launch_bounds__(256) k_gemm_qkv(const __bf16* __restrict__ Wq,
                                                  const __bf16* __restrict__ X,
                                                  float* __restrict__ Y) {
  const int b = blockIdx.z;
  const int m0 = blockIdx.y * 128;
  const int n0 = blockIdx.x * 128;
  gemm128x128_bf16(Wq + (size_t)m0 * kC, kC,
                   X + (size_t)b * kC * kN + n0, kN,
                   Y + ((size_t)b * kQKV + m0) * kN + n0, kN, kC);
}

// ---------------------------------------------------------------- kernel 4:
// depthwise 3x3, zero pad
__global__ void __launch_bounds__(256) k_dwconv(const float* __restrict__ in,
                                                const float* __restrict__ wdw,
                                                float* __restrict__ out) {
  const int p = blockIdx.x * 256 + threadIdx.x;
  const int ch = blockIdx.y, b = blockIdx.z;
  const int h = p / 96, w = p - h * 96;
  const size_t base = ((size_t)b * kQKV + ch) * kN;
  float acc = 0.f;
#pragma unroll
  for (int dh = 0; dh < 3; ++dh) {
    int hh = h + dh - 1;
    if (hh < 0 || hh >= 96) continue;
#pragma unroll
    for (int dw = 0; dw < 3; ++dw) {
      int ww = w + dw - 1;
      if (ww < 0 || ww >= 96) continue;
      acc += in[base + (size_t)hh * 96 + ww] * wdw[ch * 9 + dh * 3 + dw];
    }
  }
  out[base + p] = acc;
}

// ---------------------------------------------------------------- kernel 5:
// stable bitonic argsort of v rows (9216 -> padded 16384) + gather q/k
__global__ void __launch_bounds__(1024) k_sort_v(const float* __restrict__ qkv,
                                                 int* __restrict__ idxv,
                                                 float* __restrict__ q1s,
                                                 float* __restrict__ k1s,
                                                 float* __restrict__ q2s,
                                                 float* __restrict__ k2s,
                                                 float* __restrict__ vs) {
  constexpr int SN = 16384;
  __shared__ float sv[SN];
  __shared__ unsigned short si[SN];
  const int bc = blockIdx.x;
  const int b = bc >> 8, c = bc & 255;
  const float* vrow = qkv + ((size_t)b * kQKV + 4 * kC + c) * kN;
  for (int i = threadIdx.x; i < SN; i += 1024) {
    sv[i] = (i < kN) ? vrow[i] : 3.4e38f;
    si[i] = (unsigned short)i;
  }
  __syncthreads();
  for (int k = 2; k <= SN; k <<= 1) {
    for (int j = k >> 1; j > 0; j >>= 1) {
      for (int i = threadIdx.x; i < SN; i += 1024) {
        const int ix = i ^ j;
        if (ix > i) {
          const bool up = ((i & k) == 0);
          float a = sv[i], bb = sv[ix];
          unsigned short ia = si[i], ibn = si[ix];
          const bool agtb = (a > bb) || (a == bb && ia > ibn);  // idx tiebreak = stable
          if (agtb == up) { sv[i] = bb; sv[ix] = a; si[i] = ibn; si[ix] = ia; }
        }
      }
      __syncthreads();
    }
  }
  const float* base = qkv + (size_t)b * kQKV * kN;
  for (int i = threadIdx.x; i < kN; i += 1024) {
    const int s = si[i];
    const size_t o = ((size_t)b * kC + c) * kN + i;
    idxv[o] = s;
    q1s[o] = base[((size_t)0 * kC + c) * kN + s];
    k1s[o] = base[((size_t)1 * kC + c) * kN + s];
    q2s[o] = base[((size_t)2 * kC + c) * kN + s];
    k2s[o] = base[((size_t)3 * kC + c) * kN + s];
    vs[o]  = sv[i];
  }
}

// ---------------------------------------------------------------- kernel 6:
// build head matrices (reshape + L2 row normalize for q/k), bf16 out
__global__ void __launch_bounds__(128) k_build_heads(
    const float* __restrict__ q1s, const float* __restrict__ k1s,
    const float* __restrict__ q2s, const float* __restrict__ k2s,
    const float* __restrict__ vs, __bf16* qh1, __bf16* kh1, __bf16* vh1,
    __bf16* qh2, __bf16* kh2, __bf16* vh2) {
  __shared__ float red[128];
  const int row = blockIdx.x;        // ((b*8+h)*256 + m)
  const int t = blockIdx.y;          // 0..5: q1,k1,v1,q2,k2,v2
  const int bh = row >> 8;
  const int m = row & 255;
  const int h = bh & 7;
  const int b = bh >> 3;
  const int cp = m >> 3, f = m & 7;
  const int c = h * 32 + cp;
  const int branch = t / 3, kind = t % 3;
  const float* src = (kind == 0) ? (branch ? q2s : q1s)
                   : (kind == 1) ? (branch ? k2s : k1s) : vs;
  __bf16* dst = (kind == 0) ? (branch ? qh2 : qh1)
              : (kind == 1) ? (branch ? kh2 : kh1) : (branch ? vh2 : vh1);
  const size_t srow = ((size_t)b * kC + c) * kN;
  float v[9];
  float ss = 0.f;
#pragma unroll
  for (int i = 0; i < 9; ++i) {
    const int d = threadIdx.x + i * 128;
    const int n = branch ? (d * 8 + f) : (f * kHW + d);
    v[i] = src[srow + n];
    ss += v[i] * v[i];
  }
  red[threadIdx.x] = ss;
  __syncthreads();
  for (int s = 64; s > 0; s >>= 1) {
    if (threadIdx.x < s) red[threadIdx.x] += red[threadIdx.x + s];
    __syncthreads();
  }
  float scale = 1.f;
  if (kind < 2) scale = 1.f / fmaxf(sqrtf(red[0]), 1e-12f);
  const size_t drow = ((size_t)bh * kM + m) * kHW;
#pragma unroll
  for (int i = 0; i < 9; ++i)
    dst[drow + threadIdx.x + i * 128] = (__bf16)(v[i] * scale);
}

// ---------------------------------------------------------------- kernel 7:
// attention scores: qh*khT (WMMA) fused with temp scale, exp, e/(sum+1), bf16
__global__ void __launch_bounds__(256) k_attn_scores(
    const __bf16* __restrict__ qh1, const __bf16* __restrict__ kh1,
    const __bf16* __restrict__ qh2, const __bf16* __restrict__ kh2,
    const float* __restrict__ temperature, __bf16* __restrict__ attn) {
  __shared__ __bf16 lA[128 * 32];
  __shared__ __bf16 lB[256 * 32];
  const int branch = blockIdx.z;
  const int bh = blockIdx.y;
  const int h = bh & 7;
  const int m0 = blockIdx.x * 128;
  const __bf16* qh = (branch ? qh2 : qh1) + ((size_t)bh * kM + m0) * kHW;
  const __bf16* kh = (branch ? kh2 : kh1) + (size_t)bh * kM * kHW;
  const int wave = threadIdx.x >> 5;
  const int lane = threadIdx.x & 31;
  v8f acc[16];
#pragma unroll
  for (int t = 0; t < 16; ++t) acc[t] = zero8();
  for (int k0 = 0; k0 < kHW; k0 += 32) {
    __syncthreads();
    if (threadIdx.x < 32) {
      tdm_load_2d_bf16(lds_byte_off(lA), qh + k0, 32u, 128u, (unsigned)kHW);
      tdm_load_2d_bf16(lds_byte_off(lB), kh + k0, 32u, 256u, (unsigned)kHW);
      __builtin_amdgcn_s_wait_tensorcnt(0);
    }
    __syncthreads();
    v16bf af = load_a_frag(lA + (size_t)(wave * 16) * 32, 32);
#pragma unroll
    for (int t = 0; t < 16; ++t) {
      // kh tile is n-major (n x k) == B^T storage: same striping as an A frag
      v16bf bf = load_a_frag(lB + (size_t)(t * 16) * 32, 32);
      acc[t] = wmma_bf16(af, bf, acc[t]);
    }
  }
  const float ts = temperature[h];
  float e[16][8];
  float rs[8];
#pragma unroll
  for (int r = 0; r < 8; ++r) rs[r] = 0.f;
#pragma unroll
  for (int t = 0; t < 16; ++t)
#pragma unroll
    for (int r = 0; r < 8; ++r) {
      e[t][r] = __expf(acc[t][r] * ts);
      rs[r] += e[t][r];
    }
#pragma unroll
  for (int r = 0; r < 8; ++r) {
    rs[r] += __shfl_xor(rs[r], 1, 32);
    rs[r] += __shfl_xor(rs[r], 2, 32);
    rs[r] += __shfl_xor(rs[r], 4, 32);
    rs[r] += __shfl_xor(rs[r], 8, 32);
  }
  const int col = lane & 15;
  const int rb = (lane >> 4) * 8;
  __bf16* dst = attn + (((size_t)branch * 32 + bh) * kM + (m0 + wave * 16)) * kM;
#pragma unroll
  for (int t = 0; t < 16; ++t)
#pragma unroll
    for (int r = 0; r < 8; ++r)
      dst[(size_t)(r + rb) * kM + t * 16 + col] =
          (__bf16)(e[t][r] / (rs[r] + 1.0f));
}

// ---------------------------------------------------------------- kernel 8:
// out = attn (256x256) x vh (256x1152), WMMA
__global__ void __launch_bounds__(256) k_attn_out(const __bf16* __restrict__ attn,
                                                  const __bf16* __restrict__ vh1,
                                                  const __bf16* __restrict__ vh2,
                                                  float* __restrict__ outh) {
  const int z = blockIdx.z;
  const int branch = z >> 5;
  const int bh = z & 31;
  const int m0 = blockIdx.y * 128;
  const int n0 = blockIdx.x * 128;
  const __bf16* A = attn + (((size_t)branch * 32 + bh) * kM + m0) * kM;
  const __bf16* Bv = (branch ? vh2 : vh1) + (size_t)bh * kM * kHW + n0;
  float* C = outh + (((size_t)branch * 32 + bh) * kM + m0) * kHW + n0;
  gemm128x128_bf16(A, kM, Bv, kHW, C, kHW, kM);
}

// ---------------------------------------------------------------- kernel 9:
// un-reshape both branches, inverse-permute by idx, multiply -> bf16
__global__ void __launch_bounds__(256) k_combine(const float* __restrict__ outh,
                                                 const int* __restrict__ idxv,
                                                 __bf16* __restrict__ prod) {
  const int bc = blockIdx.x;
  const int b = bc >> 8, c = bc & 255;
  const int h = c >> 5, cp = c & 31;
  const size_t rb = ((size_t)b * kC + c) * kN;
  const float* oh1 = outh + (size_t)(b * 8 + h) * kM * kHW;
  const float* oh2 = outh + (size_t)(32 + b * 8 + h) * kM * kHW;
  for (int n = threadIdx.x; n < kN; n += 256) {
    const int f1 = n / kHW, d1 = n - f1 * kHW;
    const int f2 = n & 7, d2 = n >> 3;
    const float v1 = oh1[(size_t)(cp * 8 + f1) * kHW + d1];
    const float v2 = oh2[(size_t)(cp * 8 + f2) * kHW + d2];
    const int p = idxv[rb + n];
    prod[rb + p] = (__bf16)(v1 * v2);
  }
}

// ---------------------------------------------------------------- kernel 10:
// projection 1x1 conv as GEMM (256x256) x (256x9216)
__global__ void __launch_bounds__(256) k_proj(const __bf16* __restrict__ Wp,
                                              const __bf16* __restrict__ P,
                                              float* __restrict__ Y) {
  const int b = blockIdx.z;
  const int m0 = blockIdx.y * 128;
  const int n0 = blockIdx.x * 128;
  gemm128x128_bf16(Wp + (size_t)m0 * kC, kC,
                   P + (size_t)b * kC * kN + n0, kN,
                   Y + ((size_t)b * kC + m0) * kN + n0, kN, kC);
}

// ---------------------------------------------------------------- kernel 11:
// final inverse scatters on first half
__global__ void __launch_bounds__(256) k_final(const float* __restrict__ tmp,
                                               const int* __restrict__ idx_h,
                                               const int* __restrict__ idx_w,
                                               float* __restrict__ out) {
  const int c = blockIdx.x, b = blockIdx.y;
  const size_t base = ((size_t)b * kC + c) * kN;
  if (c >= kHalf) {
    for (int p = threadIdx.x; p < kN; p += 256) out[base + p] = tmp[base + p];
  } else {
    const size_t ib = ((size_t)b * kHalf + c) * kN;
    for (int p = threadIdx.x; p < kN; p += 256) {
      const int h = p / 96;
      const int iw = idx_w[ib + p];
      const int ih = idx_h[ib + (size_t)h * 96 + iw];
      out[base + (size_t)ih * 96 + iw] = tmp[base + p];
    }
  }
}

// ---------------------------------------------------------------- launch
extern "C" void kernel_launch(void* const* d_in, const int* in_sizes, int n_in,
                              void* d_out, int out_size, void* d_ws,
                              size_t ws_size, hipStream_t stream) {
  (void)in_sizes; (void)n_in; (void)out_size; (void)ws_size;
  const float* x      = (const float*)d_in[0];
  const float* w_qkv  = (const float*)d_in[1];
  const float* w_dw   = (const float*)d_in[2];
  const float* w_proj = (const float*)d_in[3];
  const float* temp   = (const float*)d_in[4];
  float* out = (float*)d_out;

  char* ws = (char*)d_ws;
  size_t off = 0;
  auto carve = [&](size_t bytes) -> void* {
    void* p = ws + off;
    off += (bytes + 255) & ~(size_t)255;
    return p;
  };
  __bf16* x_bf  = (__bf16*)carve((size_t)kB * kC * kN * 2);
  int* idx_h    = (int*)carve((size_t)kB * kHalf * kN * 4);
  int* idx_w    = (int*)carve((size_t)kB * kHalf * kN * 4);
  __bf16* wq_bf = (__bf16*)carve((size_t)kQKV * kC * 2);
  __bf16* wp_bf = (__bf16*)carve((size_t)kC * kC * 2);
  float* qkv_l  = (float*)carve((size_t)kB * kQKV * kN * 4);
  float* qkv    = (float*)carve((size_t)kB * kQKV * kN * 4);
  int* idxv     = (int*)carve((size_t)kB * kC * kN * 4);
  float* q1s    = (float*)carve((size_t)kB * kC * kN * 4);
  float* k1s    = (float*)carve((size_t)kB * kC * kN * 4);
  float* q2s    = (float*)carve((size_t)kB * kC * kN * 4);
  float* k2s    = (float*)carve((size_t)kB * kC * kN * 4);
  float* vs     = (float*)carve((size_t)kB * kC * kN * 4);
  const size_t headB = (size_t)kB * kHeads * kM * kHW;
  __bf16* qh1 = (__bf16*)carve(headB * 2);
  __bf16* kh1 = (__bf16*)carve(headB * 2);
  __bf16* vh1 = (__bf16*)carve(headB * 2);
  __bf16* qh2 = (__bf16*)carve(headB * 2);
  __bf16* kh2 = (__bf16*)carve(headB * 2);
  __bf16* vh2 = (__bf16*)carve(headB * 2);
  __bf16* attn = (__bf16*)carve((size_t)2 * kB * kHeads * kM * kM * 2);
  float* outh  = (float*)carve((size_t)2 * headB * 4);
  __bf16* prod = (__bf16*)carve((size_t)kB * kC * kN * 2);
  float* tmpo  = (float*)carve((size_t)kB * kC * kN * 4);

  k_sort_hw<<<dim3(kC, kB), 96, 0, stream>>>(x, x_bf, idx_h, idx_w);
  k_cvt_w<<<dim3(kQKV), 256, 0, stream>>>(w_qkv, w_proj, wq_bf, wp_bf);
  k_gemm_qkv<<<dim3(kN / 128, kQKV / 128, kB), 256, 0, stream>>>(wq_bf, x_bf, qkv_l);
  k_dwconv<<<dim3(kN / 256, kQKV, kB), 256, 0, stream>>>(qkv_l, w_dw, qkv);
  k_sort_v<<<dim3(kB * kC), 1024, 0, stream>>>(qkv, idxv, q1s, k1s, q2s, k2s, vs);
  k_build_heads<<<dim3(kB * kHeads * kM, 6), 128, 0, stream>>>(
      q1s, k1s, q2s, k2s, vs, qh1, kh1, vh1, qh2, kh2, vh2);
  k_attn_scores<<<dim3(2, kB * kHeads, 2), 256, 0, stream>>>(qh1, kh1, qh2, kh2,
                                                             temp, attn);
  k_attn_out<<<dim3(kHW / 128, 2, 2 * kB * kHeads), 256, 0, stream>>>(attn, vh1,
                                                                      vh2, outh);
  k_combine<<<dim3(kB * kC), 256, 0, stream>>>(outh, idxv, prod);
  k_proj<<<dim3(kN / 128, kC / 128, kB), 256, 0, stream>>>(wp_bf, prod, tmpo);
  k_final<<<dim3(kC, kB), 256, 0, stream>>>(tmpo, idx_h, idx_w, out);
}
